// OpenAIPrivacyFilterMLP_11974368821460
// MI455X (gfx1250) — compile-verified, matrix operand order checked by reading, not run on hardware
//
#include <hip/hip_runtime.h>
#include <cstdint>

#define ALPHA_C 1.702f
#define LIMIT_C 7.0f
#define NT 4096      // tokens (B*S)
#define NH 2048      // hidden
#define ND 2048      // intermediate
#define NE 16        // experts
#define NK 4         // top-k
#define LDSA 72      // LDS row stride in bf16 elems (144B: 16B aligned, conflict-free)

typedef __attribute__((ext_vector_type(8)))  __bf16 v8bf;
typedef __attribute__((ext_vector_type(16))) __bf16 v16bf;
typedef __attribute__((ext_vector_type(8)))  float  v8f;

__device__ __forceinline__ unsigned short f2bfu(float f) {
  unsigned int u = __builtin_bit_cast(unsigned int, f);
  u += 0x7FFFu + ((u >> 16) & 1u);           // round-to-nearest-even
  return (unsigned short)(u >> 16);
}

__device__ __forceinline__ float fast_rcp(float x) {
#if __has_builtin(__builtin_amdgcn_rcpf)
  return __builtin_amdgcn_rcpf(x);
#else
  return 1.f / x;
#endif
}

// Assemble a 16-bit WMMA operand: elements 0..7 = K {kb..kb+7}, 8..15 = K {kb+16..kb+23}
__device__ __forceinline__ v16bf ld_ab(const unsigned short* p) {
  v8bf lo = *(const v8bf*)(p);
  v8bf hi = *(const v8bf*)(p + 16);
  return __builtin_shufflevector(lo, hi, 0,1,2,3,4,5,6,7,8,9,10,11,12,13,14,15);
}

#define WMMA_BF16(A, B, C) \
  __builtin_amdgcn_wmma_f32_16x16x32_bf16(false, (A), false, (B), (short)0, (C), false, false)

// Async global->LDS 16B copy via CDNA5 GLOBAL_LOAD_ASYNC_TO_LDS_B128 (ASYNCcnt-tracked).
// Inline asm sidesteps toolchain-specific builtin signatures (bridge-doc recommended).
__device__ __forceinline__ void async_cp16(const unsigned short* g, unsigned short* l) {
  unsigned lds_off = (unsigned)(uintptr_t)l;                // LDS aperture: low 32 bits = offset
  unsigned long long ga = (unsigned long long)(uintptr_t)g;
  asm volatile("global_load_async_to_lds_b128 %0, %1, off"
               :: "v"(lds_off), "v"(ga)
               : "memory");
}
__device__ __forceinline__ void async_wait0() {
#if __has_builtin(__builtin_amdgcn_s_wait_asynccnt)
  __builtin_amdgcn_s_wait_asynccnt(0);
#else
  asm volatile("s_wait_asynccnt 0x0" ::: "memory");
#endif
}

// ---------------- Router: logits, top-4, softmax/4 ----------------
__global__ __launch_bounds__(256) void moe_router(
    const float* __restrict__ x, const float* __restrict__ rw,
    const float* __restrict__ rb, int* __restrict__ topIdx,
    float* __restrict__ topScore, float* __restrict__ scoresOut) {
  __shared__ float xs[NH];
  __shared__ float lg[NE];
  const int t = blockIdx.x, tid = threadIdx.x;
  const float* xt = x + (size_t)t * NH;
  float4 a0 = *(const float4*)(xt + tid * 8);
  float4 a1 = *(const float4*)(xt + tid * 8 + 4);
  *(float4*)(xs + tid * 8) = a0;
  *(float4*)(xs + tid * 8 + 4) = a1;
  __syncthreads();
  const int e = tid >> 4, l = tid & 15;
  const float* w = rw + (size_t)e * NH;
  float p = 0.f;
  for (int j = l; j < NH; j += 16) p += xs[j] * w[j];
  for (int s = 8; s; s >>= 1) p += __shfl_xor(p, s, 32);
  if (l == 0) lg[e] = p + rb[e];
  __syncthreads();
  if (tid == 0) {
    float v[NE];
#pragma unroll
    for (int i = 0; i < NE; ++i) v[i] = lg[i];
    int ti[NK]; float tv[NK];
#pragma unroll
    for (int k = 0; k < NK; ++k) {
      float best = -3.4e38f; int bi = 0;
#pragma unroll
      for (int i = 0; i < NE; ++i) if (v[i] > best) { best = v[i]; bi = i; }
      ti[k] = bi; tv[k] = best; v[bi] = -3.4e38f;
    }
    float m = tv[0], s = 0.f, ex[NK];
#pragma unroll
    for (int k = 0; k < NK; ++k) { ex[k] = __expf(tv[k] - m); s += ex[k]; }
    float inv = 1.f / (s * (float)NK);
#pragma unroll
    for (int k = 0; k < NK; ++k) {
      float sc = ex[k] * inv;
      topIdx[t * NK + k] = ti[k];
      topScore[t * NK + k] = sc;
      scoresOut[t * NK + k] = sc;
    }
  }
}

// ---------------- Compaction ----------------
__global__ void moe_zero16(int* counts, int* cursors) {
  int i = threadIdx.x;
  if (i < NE) { counts[i] = 0; cursors[i] = 0; }
}
__global__ void moe_count(const int* __restrict__ topIdx, int* counts) {
  int t = blockIdx.x * 256 + threadIdx.x;
  if (t < NT) {
#pragma unroll
    for (int k = 0; k < NK; ++k) atomicAdd(&counts[topIdx[t * NK + k]], 1);
  }
}
__global__ void moe_scan(const int* __restrict__ counts, int* offsets) {
  if (threadIdx.x == 0) {
    int s = 0;
    for (int e = 0; e < NE; ++e) { offsets[e] = s; s += counts[e]; }
    offsets[NE] = s;
  }
}
__global__ void moe_scatter(const int* __restrict__ topIdx,
                            const float* __restrict__ topScore,
                            const int* __restrict__ offsets, int* cursors,
                            int* __restrict__ rowTok, float* __restrict__ rowW) {
  int t = blockIdx.x * 256 + threadIdx.x;
  if (t < NT) {
#pragma unroll
    for (int k = 0; k < NK; ++k) {
      int e = topIdx[t * NK + k];
      int slot = atomicAdd(&cursors[e], 1);
      int r = offsets[e] + slot;
      rowTok[r] = t * NK + k;                          // packed (token, k-slot)
      rowW[r] = (float)NK * topScore[t * NK + k];      // folds final *TOP_K
    }
  }
}

// ---------------- GEMM1: x @ Wgu, fused clamp + SiLU-gate, bf16 out ----------------
// block tile M=128 x N=64 gated cols; K-step 64 (two WMMA slices per stage)
// 8 waves as 4(M) x 2(N); each wave 32x32 for gate AND up (8 v8f accumulators)
__global__ __launch_bounds__(256) void moe_gemm1(
    const float* __restrict__ x, const float* __restrict__ gup,
    const float* __restrict__ gbias, const int* __restrict__ offsets,
    const int* __restrict__ rowTok, unsigned short* __restrict__ gatedBuf) {
  const int e = blockIdx.z;
  const int off = offsets[e], cnt = offsets[e + 1] - off;
  const int m0 = blockIdx.y * 128;
  if (m0 >= cnt) return;
  const int n0 = blockIdx.x * 64;

  __shared__ unsigned short sA[128 * LDSA];
  __shared__ unsigned short sBg[64 * LDSA];
  __shared__ unsigned short sBu[64 * LDSA];
  __shared__ int sTok[128];

  const int tid = threadIdx.x;
  if (tid < 128) {
    int gr = m0 + tid; if (gr >= cnt) gr = cnt - 1;
    sTok[tid] = rowTok[off + gr] >> 2;
  }
  const int lane = tid & 31, hf = lane >> 4, l16 = lane & 15;
  const int w = tid >> 5, wm = w >> 1, wn = w & 1;
  const int kb = hf * 8;
  const int tr = tid >> 2, tc = (tid & 3) * 8;

  v8f vzero = {};
  v8f accg[2][2], accu[2][2];
#pragma unroll
  for (int i = 0; i < 2; ++i)
#pragma unroll
    for (int j = 0; j < 2; ++j) { accg[i][j] = vzero; accu[i][j] = vzero; }
  __syncthreads();

  for (int k0 = 0; k0 < NH; k0 += 64) {
    // stage A: gather token rows, fp32 -> bf16 (128 x 64)
#pragma unroll
    for (int rep = 0; rep < 2; ++rep) {
      int i = tr + rep * 64;
      const float* xp = x + (size_t)sTok[i] * NH + k0;
#pragma unroll
      for (int h = 0; h < 2; ++h) {
        int c = tc + h * 32;
        float4 f0 = *(const float4*)(xp + c);
        float4 f1 = *(const float4*)(xp + c + 4);
        union { unsigned short u[8]; uint4 v; } pk;
        pk.u[0] = f2bfu(f0.x); pk.u[1] = f2bfu(f0.y); pk.u[2] = f2bfu(f0.z); pk.u[3] = f2bfu(f0.w);
        pk.u[4] = f2bfu(f1.x); pk.u[5] = f2bfu(f1.y); pk.u[6] = f2bfu(f1.z); pk.u[7] = f2bfu(f1.w);
        *(uint4*)&sA[i * LDSA + c] = pk.v;
      }
    }
    // stage B transposed (gate + up panels, 64k x 64n each)
#pragma unroll
    for (int q = 0; q < 4; ++q) {
      int idx = q * 256 + tid;
      int k = idx & 63, n = (idx >> 6) * 4;
      const float* bp = gup + ((size_t)e * NH + (k0 + k)) * (2 * ND) + n0 + n;
      __builtin_prefetch(bp + 64 * 2 * ND, 0, 1);
      float4 g4 = *(const float4*)bp;
      float4 u4 = *(const float4*)(bp + ND);
      sBg[(n + 0) * LDSA + k] = f2bfu(g4.x); sBg[(n + 1) * LDSA + k] = f2bfu(g4.y);
      sBg[(n + 2) * LDSA + k] = f2bfu(g4.z); sBg[(n + 3) * LDSA + k] = f2bfu(g4.w);
      sBu[(n + 0) * LDSA + k] = f2bfu(u4.x); sBu[(n + 1) * LDSA + k] = f2bfu(u4.y);
      sBu[(n + 2) * LDSA + k] = f2bfu(u4.z); sBu[(n + 3) * LDSA + k] = f2bfu(u4.w);
    }
    __syncthreads();
#pragma unroll
    for (int kk = 0; kk < 64; kk += 32) {
      int kc = kk + kb;
      v16bf a0 = ld_ab(&sA[(wm * 32 + 0 + l16) * LDSA + kc]);
      v16bf a1 = ld_ab(&sA[(wm * 32 + 16 + l16) * LDSA + kc]);
      v16bf g0 = ld_ab(&sBg[(wn * 32 + 0 + l16) * LDSA + kc]);
      v16bf g1 = ld_ab(&sBg[(wn * 32 + 16 + l16) * LDSA + kc]);
      v16bf u0 = ld_ab(&sBu[(wn * 32 + 0 + l16) * LDSA + kc]);
      v16bf u1 = ld_ab(&sBu[(wn * 32 + 16 + l16) * LDSA + kc]);
      accg[0][0] = WMMA_BF16(a0, g0, accg[0][0]);
      accg[0][1] = WMMA_BF16(a0, g1, accg[0][1]);
      accg[1][0] = WMMA_BF16(a1, g0, accg[1][0]);
      accg[1][1] = WMMA_BF16(a1, g1, accg[1][1]);
      accu[0][0] = WMMA_BF16(a0, u0, accu[0][0]);
      accu[0][1] = WMMA_BF16(a0, u1, accu[0][1]);
      accu[1][0] = WMMA_BF16(a1, u0, accu[1][0]);
      accu[1][1] = WMMA_BF16(a1, u1, accu[1][1]);
    }
    __syncthreads();
  }
  // fused epilogue: bias, clamps, glu = gate*sigmoid(alpha*gate), gated=(up+1)*glu
#pragma unroll
  for (int sm = 0; sm < 2; ++sm)
#pragma unroll
    for (int sn = 0; sn < 2; ++sn)
#pragma unroll
      for (int r = 0; r < 8; ++r) {
        int m = wm * 32 + sm * 16 + r + hf * 8;
        int gr = m0 + m;
        if (gr < cnt) {
          int ng = n0 + wn * 32 + sn * 16 + l16;
          float gate = accg[sm][sn][r] + gbias[(size_t)e * (2 * ND) + ng];
          float up   = accu[sm][sn][r] + gbias[(size_t)e * (2 * ND) + ND + ng];
          gate = fminf(gate, LIMIT_C);
          up = fminf(fmaxf(up, -LIMIT_C), LIMIT_C);
          float glu = gate * fast_rcp(1.f + __expf(-ALPHA_C * gate));
          gatedBuf[(size_t)(off + gr) * ND + ng] = f2bfu((up + 1.f) * glu);
        }
      }
}

// ---------------- GEMM2: gated @ Wd + bd, scaled by 4*score, slot store ----------------
// block tile M=128 x N=128, K-step 64; 8 waves as 2(M) x 4(N); each wave 64x32
__global__ __launch_bounds__(256) void moe_gemm2(
    const unsigned short* __restrict__ gatedBuf, const float* __restrict__ wd,
    const float* __restrict__ dbias, const int* __restrict__ offsets,
    const int* __restrict__ rowTok, const float* __restrict__ rowW,
    float* __restrict__ outBuf) {
  const int e = blockIdx.z;
  const int off = offsets[e], cnt = offsets[e + 1] - off;
  const int m0 = blockIdx.y * 128;
  if (m0 >= cnt) return;
  const int n0 = blockIdx.x * 128;

  __shared__ unsigned short sA[128 * LDSA];
  __shared__ unsigned short sB[128 * LDSA];

  const int tid = threadIdx.x, lane = tid & 31, hf = lane >> 4, l16 = lane & 15;
  const int w = tid >> 5, wm = w >> 2, wn = w & 3;
  const int kb = hf * 8;
  const int tr = tid >> 2, tc = (tid & 3) * 8;

  v8f vzero = {};
  v8f acc[4][2];
#pragma unroll
  for (int i = 0; i < 4; ++i)
#pragma unroll
    for (int j = 0; j < 2; ++j) acc[i][j] = vzero;

  for (int k0 = 0; k0 < ND; k0 += 64) {
    // stage A: already bf16; async straight-to-LDS 16B chunks (ASYNCcnt path)
#pragma unroll
    for (int rep = 0; rep < 2; ++rep) {
      int i = tr + rep * 64;
      int gr = m0 + i; if (gr >= cnt) gr = cnt - 1;
      const unsigned short* gp = gatedBuf + (size_t)(off + gr) * ND + k0;
#pragma unroll
      for (int h = 0; h < 2; ++h) {
        int c = tc + h * 32;
        async_cp16(gp + c, &sA[i * LDSA + c]);
      }
    }
    // stage B transposed (fp32 -> bf16), 64k x 128n
#pragma unroll
    for (int q = 0; q < 8; ++q) {
      int idx = q * 256 + tid;
      int k = idx & 63, n = (idx >> 6) * 4;
      const float* bp = wd + ((size_t)e * ND + (k0 + k)) * NH + n0 + n;
      __builtin_prefetch(bp + 64 * NH, 0, 1);
      float4 b4 = *(const float4*)bp;
      sB[(n + 0) * LDSA + k] = f2bfu(b4.x); sB[(n + 1) * LDSA + k] = f2bfu(b4.y);
      sB[(n + 2) * LDSA + k] = f2bfu(b4.z); sB[(n + 3) * LDSA + k] = f2bfu(b4.w);
    }
    async_wait0();
    __syncthreads();
#pragma unroll
    for (int kk = 0; kk < 64; kk += 32) {
      int kc = kk + kb;
      v16bf a[4], b[2];
#pragma unroll
      for (int sm = 0; sm < 4; ++sm) a[sm] = ld_ab(&sA[(wm * 64 + sm * 16 + l16) * LDSA + kc]);
#pragma unroll
      for (int sn = 0; sn < 2; ++sn) b[sn] = ld_ab(&sB[(wn * 32 + sn * 16 + l16) * LDSA + kc]);
#pragma unroll
      for (int sm = 0; sm < 4; ++sm)
#pragma unroll
        for (int sn = 0; sn < 2; ++sn) acc[sm][sn] = WMMA_BF16(a[sm], b[sn], acc[sm][sn]);
    }
    __syncthreads();
  }
#pragma unroll
  for (int sm = 0; sm < 4; ++sm)
#pragma unroll
    for (int sn = 0; sn < 2; ++sn)
#pragma unroll
      for (int r = 0; r < 8; ++r) {
        int m = wm * 64 + sm * 16 + r + hf * 8;
        int gr = m0 + m;
        if (gr < cnt) {
          int ng = n0 + wn * 32 + sn * 16 + l16;
          float v = acc[sm][sn][r] + dbias[(size_t)e * NH + ng];
          int pk = rowTok[off + gr];                    // t*4 + kslot
          outBuf[(size_t)pk * NH + ng] = v * rowW[off + gr];
        }
      }
}

// ---------------- Combine: y[t] = sum over 4 slots (deterministic, no atomics) ----------------
__global__ __launch_bounds__(256) void moe_combine(const float* __restrict__ outBuf,
                                                   float* __restrict__ y) {
  size_t idx = ((size_t)blockIdx.x * 256 + threadIdx.x) * 4;
  size_t t = idx >> 11;        // / NH
  size_t n = idx & (NH - 1);
  float4 s = {0.f, 0.f, 0.f, 0.f};
#pragma unroll
  for (int k = 0; k < NK; ++k) {
    float4 v = *(const float4*)(outBuf + ((t * NK + k) * NH) + n);
    s.x += v.x; s.y += v.y; s.z += v.z; s.w += v.w;
  }
  *(float4*)(y + idx) = s;
}

extern "C" void kernel_launch(void* const* d_in, const int* in_sizes, int n_in,
                              void* d_out, int out_size, void* d_ws, size_t ws_size,
                              hipStream_t stream) {
  const float* x     = (const float*)d_in[0];  // [B,S,H]
  const float* rw    = (const float*)d_in[1];  // [E,H]
  const float* rb    = (const float*)d_in[2];  // [E]
  const float* gup   = (const float*)d_in[3];  // [E,H,2D]
  const float* gbias = (const float*)d_in[4];  // [E,2D]
  const float* wd    = (const float*)d_in[5];  // [E,D,H]
  const float* dbias = (const float*)d_in[6];  // [E,H]

  float* y = (float*)d_out;                    // [T*H] then scores [T*K]
  float* scoresOut = y + (size_t)NT * NH;

  char* ws = (char*)d_ws;
  int*   topIdx   = (int*)ws;    ws += (size_t)NT * NK * sizeof(int);
  float* topScore = (float*)ws;  ws += (size_t)NT * NK * sizeof(float);
  int*   counts   = (int*)ws;    ws += 64;
  int*   offsets  = (int*)ws;    ws += 128;
  int*   cursors  = (int*)ws;    ws += 64;
  int*   rowTok   = (int*)ws;    ws += (size_t)NT * NK * sizeof(int);
  float* rowW     = (float*)ws;  ws += (size_t)NT * NK * sizeof(float);
  ws = (char*)(((uintptr_t)ws + 255) & ~(uintptr_t)255);
  unsigned short* gatedBuf = (unsigned short*)ws;
  ws += (size_t)NT * NK * ND * sizeof(unsigned short);
  ws = (char*)(((uintptr_t)ws + 255) & ~(uintptr_t)255);
  float* outBuf = (float*)ws;

  moe_router<<<NT, 256, 0, stream>>>(x, rw, rb, topIdx, topScore, scoresOut);
  moe_zero16<<<1, 32, 0, stream>>>(counts, cursors);
  moe_count<<<NT / 256, 256, 0, stream>>>(topIdx, counts);
  moe_scan<<<1, 32, 0, stream>>>(counts, offsets);
  moe_scatter<<<NT / 256, 256, 0, stream>>>(topIdx, topScore, offsets, cursors, rowTok, rowW);
  moe_gemm1<<<dim3(ND / 64, NT / 128, NE), 256, 0, stream>>>(x, gup, gbias, offsets, rowTok, gatedBuf);
  moe_gemm2<<<dim3(NH / 128, NT / 128, NE), 256, 0, stream>>>(gatedBuf, wd, dbias, offsets, rowTok, rowW, outBuf);
  moe_combine<<<(size_t)NT * NH / 4 / 256, 256, 0, stream>>>(outBuf, y);
}
